// LSTMAutoencoder_14766097563664
// MI455X (gfx1250) — matrix operand order repaired and checked by reading, not compiled, not run on hardware
//
#include <hip/hip_runtime.h>
#include <hip/hip_bf16.h>

typedef __attribute__((ext_vector_type(16))) _Float16 v16h;
typedef __attribute__((ext_vector_type(8)))  _Float16 v8h;
typedef __attribute__((ext_vector_type(8)))  float    v8f;

namespace lstmae {

constexpr int B = 64, T = 1024, D = 128, H = 512, G = 2048, Z = 64;
constexpr int NWG = 32, THREADS = 128;   // 32 WGs x 4 wave32 = 128 waves

// ---------------- workspace layout (bytes, all offsets 256B aligned) ----------------
constexpr size_t O_XH    = 0;                                   // x in f16, [B][T][D]
constexpr size_t O_EWIH0 = O_XH    + (size_t)2 * B * T * D;
constexpr size_t O_EWHH0 = O_EWIH0 + (size_t)2 * G * D;
constexpr size_t O_EWIH1 = O_EWHH0 + (size_t)2 * G * H;
constexpr size_t O_EWHH1 = O_EWIH1 + (size_t)2 * G * H;
constexpr size_t O_DWHH0 = O_EWHH1 + (size_t)2 * G * H;
constexpr size_t O_DWIH1 = O_DWHH0 + (size_t)2 * G * H;
constexpr size_t O_DWHH1 = O_DWIH1 + (size_t)2 * G * H;
constexpr size_t O_OPW   = O_DWHH1 + (size_t)2 * G * H;
constexpr size_t O_EB0   = O_OPW   + (size_t)2 * D * H;         // f32 bih+bhh sums
constexpr size_t O_EB1   = O_EB0   + (size_t)4 * G;
constexpr size_t O_DB0   = O_EB1   + (size_t)4 * G;
constexpr size_t O_DB1   = O_DB0   + (size_t)4 * G;
constexpr size_t O_H0A   = O_DB1   + (size_t)4 * G;             // f16 h ping-pong buffers
constexpr size_t O_H0B   = O_H0A   + (size_t)2 * B * H;
constexpr size_t O_H1A   = O_H0B   + (size_t)2 * B * H;
constexpr size_t O_H1B   = O_H1A   + (size_t)2 * B * H;
constexpr size_t O_ZL    = O_H1B   + (size_t)2 * B * H;         // latent z f32
constexpr size_t O_CNT   = O_ZL    + (size_t)4 * B * Z;         // barrier counters

// ---------------- helpers ----------------
__device__ __forceinline__ float fsig(float x)  { return 1.0f / (1.0f + __expf(-x)); }
__device__ __forceinline__ float ftanh(float x) { return 2.0f / (1.0f + __expf(-2.0f * x)) - 1.0f; }

__device__ __forceinline__ v8f splat8(float x) {
  v8f v;
#pragma unroll
  for (int i = 0; i < 8; ++i) v[i] = x;
  return v;
}

__device__ __forceinline__ v8f wmma16(v16h a, v16h b, v8f c) {
  // v_wmma_f32_16x16x32_f16: D = A(16x32 f16) * B(32x16 f16) + C(f32)
  return __builtin_amdgcn_wmma_f32_16x16x32_f16(false, a, false, b, (short)0, c, false, false);
}

// A-tile 16x32 f16 (row-major source): lanes 0-15 hold K 0-15, lanes 16-31 hold K 16-31
__device__ __forceinline__ v16h load_a_tile(const _Float16* base, long rstride, int row0, long col0, int lane) {
  const _Float16* q = base + (long)(row0 + (lane & 15)) * rstride + col0 + (long)((lane >> 4) << 4);
  v8h lo = *(const v8h*)q;
  v8h hi = *(const v8h*)(q + 8);
  v16h a;
#pragma unroll
  for (int i = 0; i < 8; ++i) { a[i] = lo[i]; a[i + 8] = hi[i]; }
  return a;
}

// B-tile 32x16 f16 from row-major weights [N][K]: lane n=L%16, 16 contiguous K at K0 + 16*(L/16)
__device__ __forceinline__ v16h load_b_tile(const _Float16* base, int rstride, int n0, int k0, int lane) {
  return *(const v16h*)(base + (long)(n0 + (lane & 15)) * rstride + k0 + ((lane >> 4) << 4));
}

struct Acc4 { v8f i, f, g, o; };

// accumulate 4 gate tiles (i,f,g,o share the A operand) over K
__device__ __forceinline__ void gemm_gates(Acc4& acc, const _Float16* A, long astride, int m0, long acol0,
                                           const _Float16* W, int wld, int j0, int K, int lane) {
#pragma unroll 2
  for (int k = 0; k < K; k += 32) {
    v16h a = load_a_tile(A, astride, m0, acol0 + k, lane);
    acc.i = wmma16(a, load_b_tile(W, wld, 0 * H + j0, k, lane), acc.i);
    acc.f = wmma16(a, load_b_tile(W, wld, 1 * H + j0, k, lane), acc.f);
    acc.g = wmma16(a, load_b_tile(W, wld, 2 * H + j0, k, lane), acc.g);
    acc.o = wmma16(a, load_b_tile(W, wld, 3 * H + j0, k, lane), acc.o);
  }
}

// LSTM cell update on the 16x16 tile; c stays in registers, h stored f16 for next-step WMMA A operand
__device__ __forceinline__ void lstm_pointwise(Acc4& g4, v8f& c, _Float16* hdst, int m0, int j0, int lane) {
  const int nl = lane & 15;
  const int rowoff = (lane >> 4) * 8;   // C/D layout: VGPR r -> M = r (+8 for upper lane half)
#pragma unroll
  for (int r = 0; r < 8; ++r) {
    float cv = fsig(g4.f[r]) * c[r] + fsig(g4.i[r]) * ftanh(g4.g[r]);
    c[r] = cv;
    float hv = fsig(g4.o[r]) * ftanh(cv);
    hdst[(long)(m0 + r + rowoff) * H + j0 + nl] = (_Float16)hv;
  }
}

// device-scope barrier: monotone counter, release stores / acquire spin, s_sleep while waiting
__device__ __forceinline__ void grid_sync(unsigned* cnt, unsigned* epoch) {
  __threadfence();
  __syncthreads();
  if (threadIdx.x == 0) {
    unsigned e = ++(*epoch);
    __hip_atomic_fetch_add(cnt, 1u, __ATOMIC_ACQ_REL, __HIP_MEMORY_SCOPE_AGENT);
    const unsigned target = e * gridDim.x;
    while (__hip_atomic_load(cnt, __ATOMIC_ACQUIRE, __HIP_MEMORY_SCOPE_AGENT) < target)
      __builtin_amdgcn_s_sleep(1);
    __threadfence();
  }
  __syncthreads();
}

} // namespace lstmae

using namespace lstmae;

// ---------------- utility kernels ----------------
__global__ void lstmae_f32_to_f16(const float* __restrict__ src, _Float16* __restrict__ dst, long n) {
  long i = (long)blockIdx.x * blockDim.x + threadIdx.x;
  long stride = (long)gridDim.x * blockDim.x;
  for (; i < n; i += stride) dst[i] = (_Float16)src[i];
}

__global__ void lstmae_bias_sum(const float* __restrict__ a, const float* __restrict__ b,
                                float* __restrict__ d, int n) {
  int i = blockIdx.x * blockDim.x + threadIdx.x;
  if (i < n) d[i] = a[i] + b[i];
}

__global__ void lstmae_init_state(float* __restrict__ hzero, unsigned* __restrict__ cnt) {
  int i = blockIdx.x * blockDim.x + threadIdx.x;
  if (i < (int)(4 * 2 * B * H / 4)) hzero[i] = 0.0f;   // 4 f16 h buffers, zeroed as f32 words
  if (i < 64) cnt[i] = 0u;
}

// z = h_enc_final @ tl_W.T + tl_b  (tiny: scalar f32)
__global__ void lstmae_latent(const _Float16* __restrict__ h1, const float* __restrict__ tlW,
                              const float* __restrict__ tlb, float* __restrict__ z_ws,
                              float* __restrict__ z_out) {
  int i = blockIdx.x * blockDim.x + threadIdx.x;
  if (i >= B * Z) return;
  int b = i / Z, u = i % Z;
  float acc = tlb[u];
  for (int k = 0; k < H; ++k) acc += (float)h1[b * H + k] * tlW[u * H + k];
  z_ws[i] = acc;
  z_out[i] = acc;
}

// init_h = tanh(z @ lh_W.T + lh_b).reshape(L,B,H) -> decoder h0/h1 (f16)
__global__ void lstmae_init_hidden(const float* __restrict__ z, const float* __restrict__ lhW,
                                   const float* __restrict__ lhb,
                                   _Float16* __restrict__ h0a, _Float16* __restrict__ h1a) {
  int i = blockIdx.x * blockDim.x + threadIdx.x;
  if (i >= 2 * B * H) return;
  int l = i / (B * H);
  int rem = i % (B * H);
  int b = rem / H, j = rem % H;
  int row = l * H + j;
  float acc = lhb[row];
  for (int k = 0; k < Z; ++k) acc += z[b * Z + k] * lhW[row * Z + k];
  _Float16 v = (_Float16)ftanh(acc);
  if (l == 0) h0a[b * H + j] = v; else h1a[b * H + j] = v;
}

// ---------------- persistent encoder: 2-layer LSTM scan over T ----------------
__global__ __launch_bounds__(THREADS, 1) void lstmae_encoder(
    const _Float16* __restrict__ xh,
    const _Float16* __restrict__ Wih0, const _Float16* __restrict__ Whh0,
    const _Float16* __restrict__ Wih1, const _Float16* __restrict__ Whh1,
    const float* __restrict__ b0, const float* __restrict__ b1,
    _Float16* __restrict__ h0a, _Float16* __restrict__ h0b,
    _Float16* __restrict__ h1a, _Float16* __restrict__ h1b,
    unsigned* __restrict__ cnt) {
  __shared__ unsigned epoch;
  if (threadIdx.x == 0) epoch = 0;
  __syncthreads();
  const int lane = threadIdx.x & 31;
  const int wave = blockIdx.x * (THREADS >> 5) + (threadIdx.x >> 5);  // 0..127
  const int m0 = (wave >> 5) << 4;    // batch tile (4 tiles of 16)
  const int j0 = (wave & 31) << 4;    // hidden tile (32 tiles of 16)
  const int nl = lane & 15;
  const float bi0 = b0[0 * H + j0 + nl], bf0 = b0[1 * H + j0 + nl];
  const float bg0 = b0[2 * H + j0 + nl], bo0 = b0[3 * H + j0 + nl];
  const float bi1 = b1[0 * H + j0 + nl], bf1 = b1[1 * H + j0 + nl];
  const float bg1 = b1[2 * H + j0 + nl], bo1 = b1[3 * H + j0 + nl];
  v8f c0 = {}, c1 = {};                       // cell state lives in registers for all 1024 steps
  _Float16 *h0rd = h0a, *h0wr = h0b, *h1rd = h1a, *h1wr = h1b;
  for (int t = 0; t < T; ++t) {
    // layer 0: gates = x_t @ Wih0^T + h0 @ Whh0^T + b
    Acc4 g0 = { splat8(bi0), splat8(bf0), splat8(bg0), splat8(bo0) };
    gemm_gates(g0, xh, (long)T * D, m0, (long)t * D, Wih0, D, j0, D, lane);
    gemm_gates(g0, h0rd, H, m0, 0, Whh0, H, j0, H, lane);
    lstm_pointwise(g0, c0, h0wr, m0, j0, lane);
    grid_sync(cnt, &epoch);
    // layer 1: gates = h0_t @ Wih1^T + h1 @ Whh1^T + b
    Acc4 g1 = { splat8(bi1), splat8(bf1), splat8(bg1), splat8(bo1) };
    gemm_gates(g1, h0wr, H, m0, 0, Wih1, H, j0, H, lane);
    gemm_gates(g1, h1rd, H, m0, 0, Whh1, H, j0, H, lane);
    lstm_pointwise(g1, c1, h1wr, m0, j0, lane);
    grid_sync(cnt, &epoch);
    _Float16* tp;
    tp = h0rd; h0rd = h0wr; h0wr = tp;
    tp = h1rd; h1rd = h1wr; h1wr = tp;
  }
  // T even -> final h1 ends in h1a
}

// ---------------- persistent decoder: zero-input 2-layer LSTM + fused output projection ----------------
__global__ __launch_bounds__(THREADS, 1) void lstmae_decoder(
    const _Float16* __restrict__ Whh0,
    const _Float16* __restrict__ Wih1, const _Float16* __restrict__ Whh1,
    const _Float16* __restrict__ opW, const float* __restrict__ opb,
    const float* __restrict__ b0, const float* __restrict__ b1,
    _Float16* __restrict__ h0a, _Float16* __restrict__ h0b,
    _Float16* __restrict__ h1a, _Float16* __restrict__ h1b,
    float* __restrict__ recon, unsigned* __restrict__ cnt) {
  __shared__ unsigned epoch;
  if (threadIdx.x == 0) epoch = 0;
  __syncthreads();
  const int lane = threadIdx.x & 31;
  const int wave = blockIdx.x * (THREADS >> 5) + (threadIdx.x >> 5);
  const int m0 = (wave >> 5) << 4;
  const int j0 = (wave & 31) << 4;
  const int nl = lane & 15;
  const float bi0 = b0[0 * H + j0 + nl], bf0 = b0[1 * H + j0 + nl];
  const float bg0 = b0[2 * H + j0 + nl], bo0 = b0[3 * H + j0 + nl];
  const float bi1 = b1[0 * H + j0 + nl], bf1 = b1[1 * H + j0 + nl];
  const float bg1 = b1[2 * H + j0 + nl], bo1 = b1[3 * H + j0 + nl];
  // projection tiles: waves 0..31 cover [64 x 128] output
  const int pm0 = ((wave >> 3) & 3) << 4;
  const int pn0 = (wave & 7) << 4;
  float pb = 0.0f;
  if (wave < 32) pb = opb[pn0 + nl];
  v8f c0 = {}, c1 = {};
  _Float16 *h0rd = h0a, *h0wr = h0b, *h1rd = h1a, *h1wr = h1b;
  for (int t = 0; t < T; ++t) {
    // layer 0: decoder input is identically zero -> gates = bias + h0 @ Whh0^T
    Acc4 g0 = { splat8(bi0), splat8(bf0), splat8(bg0), splat8(bo0) };
    gemm_gates(g0, h0rd, H, m0, 0, Whh0, H, j0, H, lane);
    lstm_pointwise(g0, c0, h0wr, m0, j0, lane);
    grid_sync(cnt, &epoch);
    // layer 1
    Acc4 g1 = { splat8(bi1), splat8(bf1), splat8(bg1), splat8(bo1) };
    gemm_gates(g1, h0wr, H, m0, 0, Wih1, H, j0, H, lane);
    gemm_gates(g1, h1rd, H, m0, 0, Whh1, H, j0, H, lane);
    lstm_pointwise(g1, c1, h1wr, m0, j0, lane);
    grid_sync(cnt, &epoch);
    // fused output projection: recon[:,t,:] = sigmoid(h1_t @ opW^T + op_b)
    // (reads h1wr just published by the barrier; next writer of h1wr is layer1 at t+2,
    //  globally ordered after every wave passed the t+1 barriers -> no extra sync needed)
    if (wave < 32) {
      v8f acc = splat8(pb);
#pragma unroll 2
      for (int k = 0; k < H; k += 32) {
        v16h a = load_a_tile(h1wr, H, pm0, 0 + k, lane);
        acc = wmma16(a, load_b_tile(opW, H, pn0, k, lane), acc);
      }
      const int rowoff = (lane >> 4) * 8;
#pragma unroll
      for (int r = 0; r < 8; ++r)
        recon[(long)(pm0 + r + rowoff) * (T * D) + (long)t * D + pn0 + nl] = fsig(acc[r]);
    }
    _Float16* tp;
    tp = h0rd; h0rd = h0wr; h0wr = tp;
    tp = h1rd; h1rd = h1wr; h1wr = tp;
  }
}

// ---------------- host-side orchestration ----------------
extern "C" void kernel_launch(void* const* d_in, const int* in_sizes, int n_in,
                              void* d_out, int out_size, void* d_ws, size_t ws_size,
                              hipStream_t stream) {
  (void)in_sizes; (void)n_in; (void)out_size; (void)ws_size;
  char* ws = (char*)d_ws;

  const float* x     = (const float*)d_in[0];
  const float* eWih0 = (const float*)d_in[1];
  const float* eWhh0 = (const float*)d_in[2];
  const float* ebih0 = (const float*)d_in[3];
  const float* ebhh0 = (const float*)d_in[4];
  const float* eWih1 = (const float*)d_in[5];
  const float* eWhh1 = (const float*)d_in[6];
  const float* ebih1 = (const float*)d_in[7];
  const float* ebhh1 = (const float*)d_in[8];
  /* d_in[9] = dec_Wih0: unused, decoder input is all-zero */
  const float* dWhh0 = (const float*)d_in[10];
  const float* dbih0 = (const float*)d_in[11];
  const float* dbhh0 = (const float*)d_in[12];
  const float* dWih1 = (const float*)d_in[13];
  const float* dWhh1 = (const float*)d_in[14];
  const float* dbih1 = (const float*)d_in[15];
  const float* dbhh1 = (const float*)d_in[16];
  const float* tlW   = (const float*)d_in[17];
  const float* tlb   = (const float*)d_in[18];
  const float* lhW   = (const float*)d_in[19];
  const float* lhb   = (const float*)d_in[20];
  const float* opW   = (const float*)d_in[21];
  const float* opb   = (const float*)d_in[22];
  float* out = (float*)d_out;

  _Float16* xh    = (_Float16*)(ws + O_XH);
  _Float16* eW0i  = (_Float16*)(ws + O_EWIH0);
  _Float16* eW0h  = (_Float16*)(ws + O_EWHH0);
  _Float16* eW1i  = (_Float16*)(ws + O_EWIH1);
  _Float16* eW1h  = (_Float16*)(ws + O_EWHH1);
  _Float16* dW0h  = (_Float16*)(ws + O_DWHH0);
  _Float16* dW1i  = (_Float16*)(ws + O_DWIH1);
  _Float16* dW1h  = (_Float16*)(ws + O_DWHH1);
  _Float16* opWh  = (_Float16*)(ws + O_OPW);
  float* eb0 = (float*)(ws + O_EB0);
  float* eb1 = (float*)(ws + O_EB1);
  float* db0 = (float*)(ws + O_DB0);
  float* db1 = (float*)(ws + O_DB1);
  _Float16* h0a = (_Float16*)(ws + O_H0A);
  _Float16* h0b = (_Float16*)(ws + O_H0B);
  _Float16* h1a = (_Float16*)(ws + O_H1A);
  _Float16* h1b = (_Float16*)(ws + O_H1B);
  float* zbuf = (float*)(ws + O_ZL);
  unsigned* cnts = (unsigned*)(ws + O_CNT);

  // 1) convert operands to f16 (WMMA operand precision; f32 accumulation in the scan)
  lstmae_f32_to_f16<<<2048, 256, 0, stream>>>(x, xh, (long)B * T * D);
  lstmae_f32_to_f16<<<512, 256, 0, stream>>>(eWih0, eW0i, (long)G * D);
  lstmae_f32_to_f16<<<1024, 256, 0, stream>>>(eWhh0, eW0h, (long)G * H);
  lstmae_f32_to_f16<<<1024, 256, 0, stream>>>(eWih1, eW1i, (long)G * H);
  lstmae_f32_to_f16<<<1024, 256, 0, stream>>>(eWhh1, eW1h, (long)G * H);
  lstmae_f32_to_f16<<<1024, 256, 0, stream>>>(dWhh0, dW0h, (long)G * H);
  lstmae_f32_to_f16<<<1024, 256, 0, stream>>>(dWih1, dW1i, (long)G * H);
  lstmae_f32_to_f16<<<1024, 256, 0, stream>>>(dWhh1, dW1h, (long)G * H);
  lstmae_f32_to_f16<<<256, 256, 0, stream>>>(opW, opWh, (long)D * H);

  // 2) fold bih+bhh once
  lstmae_bias_sum<<<8, 256, 0, stream>>>(ebih0, ebhh0, eb0, G);
  lstmae_bias_sum<<<8, 256, 0, stream>>>(ebih1, ebhh1, eb1, G);
  lstmae_bias_sum<<<8, 256, 0, stream>>>(dbih0, dbhh0, db0, G);
  lstmae_bias_sum<<<8, 256, 0, stream>>>(dbih1, dbhh1, db1, G);

  // 3) zero h state + barrier counters (fresh every launch -> graph-replay safe)
  lstmae_init_state<<<256, 256, 0, stream>>>((float*)(ws + O_H0A), cnts);

  // 4) encoder scan (persistent, register-resident cell state, 2 device barriers/step)
  lstmae_encoder<<<NWG, THREADS, 0, stream>>>(xh, eW0i, eW0h, eW1i, eW1h, eb0, eb1,
                                              h0a, h0b, h1a, h1b, cnts + 0);

  // 5) latent z -> d_out tail, and decoder initial hidden state
  lstmae_latent<<<16, 256, 0, stream>>>(h1a, tlW, tlb, zbuf, out + (long)B * T * D);
  lstmae_init_hidden<<<256, 256, 0, stream>>>(zbuf, lhW, lhb, h0a, h1a);

  // 6) decoder scan with fused sigmoid output projection straight into d_out
  lstmae_decoder<<<NWG, THREADS, 0, stream>>>(dW0h, dW1i, dW1h, opWh, opb, db0, db1,
                                              h0a, h0b, h1a, h1b, out, cnts + 32);
}
